// GINEModel_16956530884981
// MI455X (gfx1250) — compile-verified, matrix operand order checked by reading, not run on hardware
//
#include <hip/hip_runtime.h>
#include <hip/hip_bf16.h>
#include <math.h>

#define N_  100000
#define E_  1000000
#define G_  1000
#define D_  64
#define L_  5
#define T_  10
#define BN_EPS_ 1e-5f

typedef __attribute__((ext_vector_type(16))) __bf16 v16bf;
typedef __attribute__((ext_vector_type(8)))  float  v8f;

static __device__ __forceinline__ unsigned short f2bf(float f) {
    unsigned int u = __float_as_uint(f);
    unsigned int r = 0x7FFFu + ((u >> 16) & 1u);
    return (unsigned short)((u + r) >> 16);
}

// ---------------- embedding encode: out[n,d] = sum_f emb[f, idx[n,f], d] ----------------
__global__ void enc_kernel(const int* __restrict__ idx, const float* __restrict__ emb,
                           float* __restrict__ out, int n_items, int n_feats, int vocab) {
    int tid = blockIdx.x * blockDim.x + threadIdx.x;
    if (tid >= n_items * D_) return;
    int n = tid >> 6, d = tid & 63;
    float s = 0.f;
    for (int f = 0; f < n_feats; ++f) {
        int v = idx[n * n_feats + f];
        s += emb[((size_t)f * vocab + v) * D_ + d];
    }
    out[tid] = s;
}

// ---------------- weight prep: transpose + fp32 -> bf16 ----------------
__global__ void prep_weights(const float* __restrict__ W1, const float* __restrict__ W2,
                             unsigned short* __restrict__ Wt1, unsigned short* __restrict__ Wt2) {
    int tid = blockIdx.x * blockDim.x + threadIdx.x;
    const int per = L_ * 128 * 64;   // 40960
    if (tid < per) {
        int l = tid / 8192, r = tid % 8192, n = r / 64, k = r % 64;
        Wt1[tid] = f2bf(W1[l * 8192 + k * 128 + n]);    // W1: [L][64][128] -> Wt1: [L][128][64]
    } else if (tid < 2 * per) {
        int t = tid - per;
        int l = t / 8192, r = t % 8192, n = r / 128, k = r % 128;
        Wt2[t] = f2bf(W2[l * 8192 + k * 64 + n]);       // W2: [L][128][64] -> Wt2: [L][64][128]
    }
}

// ---------------- edge message + scatter-add: agg[dst] += relu(x[src] + e) ----------------
__global__ void scatter_msg(const float* __restrict__ x, const float* __restrict__ e,
                            const int* __restrict__ ei, float* __restrict__ agg) {
    int tid = blockIdx.x * blockDim.x + threadIdx.x;
    int eid = tid >> 6, d = tid & 63;
    if (eid >= E_) return;
    int s = ei[eid], dst = ei[E_ + eid];
    float v = x[(size_t)s * D_ + d] + e[(size_t)eid * D_ + d];
    v = v > 0.f ? v : 0.f;
    atomicAdd(&agg[(size_t)dst * D_ + d], v);
}

// ---------------- h0 = (1+eps)*x + agg, stored bf16 for WMMA ----------------
__global__ void combine_kernel(const float* __restrict__ x, const float* __restrict__ agg,
                               const float* __restrict__ epsp, unsigned short* __restrict__ hA) {
    int tid = blockIdx.x * blockDim.x + threadIdx.x;
    if (tid >= N_ * D_) return;
    float v = (1.f + epsp[0]) * x[tid] + agg[tid];
    hA[tid] = f2bf(v);
}

// ---------------- bf16 WMMA GEMM: C[N x NCOLS] = A[N x K] * Bt[NCOLS x K]^T + bias ----------------
// Also accumulates per-column sum / sum-of-squares into stats for BatchNorm.
template <int K, int NCOLS>
__global__ void gemm_bf16_wmma(const unsigned short* __restrict__ A,
                               const unsigned short* __restrict__ Bt,
                               const float* __restrict__ bias,
                               float* __restrict__ C,
                               float* __restrict__ stats, int nrows) {
    const int ntn = NCOLS >> 4;
    int wave = (blockIdx.x * blockDim.x + threadIdx.x) >> 5;
    int lane = threadIdx.x & 31;
    int total = (nrows >> 4) * ntn;
    if (wave >= total) return;                 // wave-uniform: EXEC stays all-ones for WMMA
    int mt = wave / ntn, nt = wave % ntn;
    int row0 = mt << 4, col0 = nt << 4;
    int half = lane >> 4;                      // 0: K 0-7/16-23, 1: K 8-15/24-31
    int l15  = lane & 15;
    int ksel = half * 8;

    const unsigned short* ap = A  + (size_t)(row0 + l15) * K + ksel;
    const unsigned short* bp = Bt + (size_t)(col0 + l15) * K + ksel;

    v8f acc = {};
#pragma unroll
    for (int k0 = 0; k0 < K; k0 += 32) {
        union { v16bf v; uint4 q[2]; } af, bfu;
        af.q[0]  = *(const uint4*)(ap + k0);
        af.q[1]  = *(const uint4*)(ap + k0 + 16);
        bfu.q[0] = *(const uint4*)(bp + k0);
        bfu.q[1] = *(const uint4*)(bp + k0 + 16);
        acc = __builtin_amdgcn_wmma_f32_16x16x32_bf16(
            false, af.v, false, bfu.v, (short)0, acc, false, false);
    }

    int nn = col0 + l15;
    float bv = bias[nn];
    int rbase = row0 + half * 8;
    float s = 0.f, sq = 0.f;
#pragma unroll
    for (int j = 0; j < 8; ++j) {
        float v = acc[j] + bv;
        C[(size_t)(rbase + j) * NCOLS + nn] = v;
        s += v; sq += v * v;
    }
    atomicAdd(&stats[nn], s);
    atomicAdd(&stats[NCOLS + nn], sq);
}

// ---------------- BN finalize: stats -> per-channel scale/shift ----------------
__global__ void bn_finalize(float* __restrict__ stats, const float* __restrict__ gamma,
                            const float* __restrict__ beta, int C, float nrows) {
    int c = threadIdx.x;
    if (c >= C) return;
    float mean = stats[c] / nrows;
    float var  = stats[C + c] / nrows - mean * mean;
    float scale = gamma[c] * rsqrtf(var + BN_EPS_);
    stats[2 * C + c] = scale;
    stats[3 * C + c] = beta[c] - mean * scale;
}

// ---------------- BN apply + ReLU, bf16 output (feeds next GEMM) ----------------
__global__ void bn_relu_bf16(const float* __restrict__ pre, const float* __restrict__ stats,
                             unsigned short* __restrict__ out, int n_elems, int C) {
    int tid = blockIdx.x * blockDim.x + threadIdx.x;
    if (tid >= n_elems) return;
    int c = tid % C;
    float v = pre[tid] * stats[2 * C + c] + stats[3 * C + c];
    v = v > 0.f ? v : 0.f;
    out[tid] = f2bf(v);
}

// ---------------- BN apply + ReLU, fp32 output (new node features x) ----------------
__global__ void bn_relu_f32(const float* __restrict__ pre, const float* __restrict__ stats,
                            float* __restrict__ out, int n_elems, int C) {
    int tid = blockIdx.x * blockDim.x + threadIdx.x;
    if (tid >= n_elems) return;
    int c = tid % C;
    float v = pre[tid] * stats[2 * C + c] + stats[3 * C + c];
    out[tid] = v > 0.f ? v : 0.f;
}

// ---------------- global mean pool (sum + count via atomics) ----------------
__global__ void pool_kernel(const float* __restrict__ x, const int* __restrict__ batch0,
                            float* __restrict__ psum, float* __restrict__ pcnt) {
    int tid = blockIdx.x * blockDim.x + threadIdx.x;
    if (tid >= N_ * D_) return;
    int n = tid >> 6, d = tid & 63;
    int g = batch0[n];
    atomicAdd(&psum[(size_t)g * D_ + d], x[tid]);
    if (d == 0) atomicAdd(&pcnt[g], 1.f);
}

// ---------------- head MLP: elu(pooled@Wp1+bp1)@Wp2+bp2 ----------------
__global__ void head_kernel(const float* __restrict__ psum, const float* __restrict__ pcnt,
                            const float* __restrict__ Wp1, const float* __restrict__ bp1,
                            const float* __restrict__ Wp2, const float* __restrict__ bp2,
                            float* __restrict__ out) {
    int g = blockIdx.x;
    __shared__ float pooled[D_];
    __shared__ float hidden[D_ / 2];
    int t = threadIdx.x;  // blockDim = 64
    float cnt = pcnt[g];
    cnt = cnt < 1.f ? 1.f : cnt;
    pooled[t] = psum[(size_t)g * D_ + t] / cnt;
    __syncthreads();
    if (t < D_ / 2) {
        float acc = bp1[t];
        for (int k = 0; k < D_; ++k) acc += pooled[k] * Wp1[k * (D_ / 2) + t];
        hidden[t] = acc > 0.f ? acc : (expf(acc) - 1.f);  // ELU(alpha=1)
    }
    __syncthreads();
    if (t < T_) {
        float acc = bp2[t];
        for (int j = 0; j < D_ / 2; ++j) acc += hidden[j] * Wp2[j * T_ + t];
        out[(size_t)g * T_ + t] = acc;
    }
}

extern "C" void kernel_launch(void* const* d_in, const int* in_sizes, int n_in,
                              void* d_out, int out_size, void* d_ws, size_t ws_size,
                              hipStream_t stream) {
    const int*   atom_x    = (const int*)d_in[0];
    const int*   edge_idx  = (const int*)d_in[1];
    const int*   edge_attr = (const int*)d_in[2];
    const int*   batch0    = (const int*)d_in[3];
    const float* atom_emb  = (const float*)d_in[4];
    const float* bond_emb  = (const float*)d_in[5];
    const float* eps       = (const float*)d_in[6];
    const float* W1        = (const float*)d_in[7];
    const float* b1        = (const float*)d_in[8];
    const float* g1        = (const float*)d_in[9];
    const float* be1       = (const float*)d_in[10];
    const float* W2        = (const float*)d_in[11];
    const float* b2        = (const float*)d_in[12];
    const float* gn        = (const float*)d_in[13];
    const float* bnb       = (const float*)d_in[14];
    const float* Wp1       = (const float*)d_in[15];
    const float* bp1       = (const float*)d_in[16];
    const float* Wp2       = (const float*)d_in[17];
    const float* bp2       = (const float*)d_in[18];
    float* out = (float*)d_out;

    // ---- carve workspace ----
    char* ws = (char*)d_ws;
    size_t off = 0;
    auto carve = [&](size_t bytes) -> void* {
        void* p = ws + off;
        off = (off + bytes + 255) & ~(size_t)255;
        return p;
    };
    float*          X    = (float*)carve((size_t)N_ * D_ * 4);
    float*          Ebuf = (float*)carve((size_t)E_ * D_ * 4);
    float*          AGG  = (float*)carve((size_t)N_ * D_ * 4);
    unsigned short* HA   = (unsigned short*)carve((size_t)N_ * D_ * 2);
    float*          PRE  = (float*)carve((size_t)N_ * 2 * D_ * 4);
    unsigned short* HB   = (unsigned short*)carve((size_t)N_ * 2 * D_ * 2);
    float*          OUT2 = (float*)carve((size_t)N_ * D_ * 4);
    unsigned short* WT1  = (unsigned short*)carve((size_t)L_ * 128 * 64 * 2);
    unsigned short* WT2  = (unsigned short*)carve((size_t)L_ * 64 * 128 * 2);
    float*          ST1  = (float*)carve(4 * 128 * 4);
    float*          ST2  = (float*)carve(4 * 64 * 4);
    float*          PSUM = (float*)carve((size_t)G_ * D_ * 4);
    float*          PCNT = (float*)carve((size_t)G_ * 4);

    const int BS = 256;
    const int gND  = (N_ * D_ + BS - 1) / BS;
    const int gED  = (E_ * D_ + BS - 1) / BS;

    // encode node + edge features
    enc_kernel<<<gND, BS, 0, stream>>>(atom_x, atom_emb, X, N_, 9, 100);
    enc_kernel<<<gED, BS, 0, stream>>>(edge_attr, bond_emb, Ebuf, E_, 3, 10);
    prep_weights<<<(2 * L_ * 8192 + BS - 1) / BS, BS, 0, stream>>>(W1, W2, WT1, WT2);

    // GINE layers
    for (int l = 0; l < L_; ++l) {
        hipMemsetAsync(AGG, 0, (size_t)N_ * D_ * 4, stream);
        scatter_msg<<<gED, BS, 0, stream>>>(X, Ebuf, edge_idx, AGG);
        combine_kernel<<<gND, BS, 0, stream>>>(X, AGG, eps + l, HA);

        hipMemsetAsync(ST1, 0, 4 * 128 * 4, stream);
        {   // GEMM1: [N x 64] x [64 x 128]  -> 6250*8 = 50000 tiles, 8 waves/block
            int tiles = (N_ / 16) * (128 / 16);
            gemm_bf16_wmma<64, 128><<<(tiles + 7) / 8, 256, 0, stream>>>(
                HA, WT1 + (size_t)l * 8192, b1 + (size_t)l * 128, PRE, ST1, N_);
        }
        bn_finalize<<<1, 128, 0, stream>>>(ST1, g1 + (size_t)l * 128, be1 + (size_t)l * 128,
                                           128, (float)N_);
        bn_relu_bf16<<<(N_ * 128 + BS - 1) / BS, BS, 0, stream>>>(PRE, ST1, HB, N_ * 128, 128);

        hipMemsetAsync(ST2, 0, 4 * 64 * 4, stream);
        {   // GEMM2: [N x 128] x [128 x 64] -> 6250*4 = 25000 tiles
            int tiles = (N_ / 16) * (64 / 16);
            gemm_bf16_wmma<128, 64><<<(tiles + 7) / 8, 256, 0, stream>>>(
                HB, WT2 + (size_t)l * 8192, b2 + (size_t)l * 64, OUT2, ST2, N_);
        }
        bn_finalize<<<1, 64, 0, stream>>>(ST2, gn + (size_t)l * 64, bnb + (size_t)l * 64,
                                          64, (float)N_);
        bn_relu_f32<<<gND, BS, 0, stream>>>(OUT2, ST2, X, N_ * D_, 64);
    }

    // pooling + head
    hipMemsetAsync(PSUM, 0, (size_t)G_ * D_ * 4, stream);
    hipMemsetAsync(PCNT, 0, (size_t)G_ * 4, stream);
    pool_kernel<<<gND, BS, 0, stream>>>(X, batch0, PSUM, PCNT);
    head_kernel<<<G_, 64, 0, stream>>>(PSUM, PCNT, Wp1, bp1, Wp2, bp2, out);

    (void)in_sizes; (void)n_in; (void)out_size; (void)ws_size;
}
